// Longformer_34050500723048
// MI455X (gfx1250) — compile-verified, matrix operand order checked by reading, not run on hardware
//
#include <hip/hip_runtime.h>
#include <math.h>

// ---------------------------------------------------------------------------
// Longformer forward for MI455X (gfx1250, wave32, WMMA).
// Big GEMMs: bf16 WMMA 16x16x32, fp32 accumulate, LDS-staged tiles with
// B stored transposed so both A and B fragments load as 2x ds_load_b128.
// Block tile 128x128, 8 waves, each wave = 32x64 strip = 2x4 wmma accums.
// Attention: streaming online-softmax (flash style), exact w.r.t. reference
// since masked entries (-1e9) contribute zero weight.
// Workspace (fp32): h,q,k,v,kg,vg,att,y (each S*DM) + f1 (S*FF) = ~151 MB.
// ---------------------------------------------------------------------------

#define S_   4096
#define DM_  768
#define FF_  3072
#define H_   12
#define D_   64
#define W_   256
#define C_   16
#define G_   8
#define L_   12
#define OUT_ 128
#define SCALE_ 0.125f // 1/sqrt(64)

typedef __attribute__((ext_vector_type(16))) __bf16 v16bf;
typedef __attribute__((ext_vector_type(8)))  __bf16 v8bf;
typedef __attribute__((ext_vector_type(8)))  float  v8f;

__device__ __forceinline__ __bf16 f2bf(float f) {
  unsigned u = __builtin_bit_cast(unsigned, f);
  unsigned r = u + 0x7FFFu + ((u >> 16) & 1u);   // round to nearest even
  unsigned short hs = (unsigned short)(r >> 16);
  return __builtin_bit_cast(__bf16, hs);
}

__device__ __forceinline__ float gelu_f(float x) {
  float x3 = x * x * x;
  return 0.5f * x * (1.0f + tanhf(0.7978845608f * (x + 0.044715f * x3)));
}

// Load one 16x32 (or 32x16 transposed) fragment row: two contiguous 16-byte
// chunks at +kb and +kb+16 elements -> 2x ds_load_b128.
__device__ __forceinline__ v16bf load_frag(const __bf16* rowbase, int kb) {
  v8bf lo = *(const v8bf*)(rowbase + kb);
  v8bf hi = *(const v8bf*)(rowbase + kb + 16);
  return __builtin_shufflevector(lo, hi, 0, 1, 2, 3, 4, 5, 6, 7,
                                 8, 9, 10, 11, 12, 13, 14, 15);
}

// ---------------------------------------------------------------------------
// WMMA GEMM: C[M,N] = act((A[M,K] @ B[K,N] + bias[N]) * scale)
// Block tile 128x128, 256 threads = 8 waves (4 along M x 2 along N),
// each wave computes a 32x64 strip = 2x4 v_wmma_f32_16x16x32_bf16.
// ---------------------------------------------------------------------------
__global__ __launch_bounds__(256)
void gemm_wmma(const float* __restrict__ A, const float* __restrict__ B,
               const float* __restrict__ bias, float* __restrict__ C,
               int M, int N, int K, float scale, int do_gelu)
{
  // 40-element rows: 80-byte stride keeps 16B alignment and spreads banks.
  __shared__ __align__(16) __bf16 As[128][40];   // A tile  [m][k], k<32
  __shared__ __align__(16) __bf16 BsT[128][40];  // B tile transposed [n][k]

  const int t    = threadIdx.x;
  const int lane = t & 31;
  const int wave = t >> 5;
  const int wm   = wave & 3;        // 0..3  (32-row strip)
  const int wn   = wave >> 2;       // 0..1  (64-col strip)
  const int bm   = blockIdx.x * 128;
  const int bn   = blockIdx.y * 128;

  v8f acc[2][4] = {};

  const int kb = (lane >> 4) << 3;  // 0 or 8 : K sub-block per half-wave

  // Staging coords (16 consecutive elements of one row per thread).
  const int ar = t >> 1;               // A row 0..127
  const int ac = (t & 1) << 4;         // A col 0 or 16
  const int bk = t >> 4;               // B row (k) 0..15  (x2 halves)
  const int bc = (t & 15) << 3;        // B col base 0..120 step 8

  for (int k0 = 0; k0 < K; k0 += 32) {
    __syncthreads();
    // ---- stage A tile 128x32: one 16-elem row chunk per thread ----
    {
      const float* ap = A + (size_t)(bm + ar) * K + k0 + ac;
      v8bf p0, p1;
      #pragma unroll
      for (int i = 0; i < 8; ++i) p0[i] = f2bf(ap[i]);
      #pragma unroll
      for (int i = 0; i < 8; ++i) p1[i] = f2bf(ap[8 + i]);
      *(v8bf*)&As[ar][ac]     = p0;
      *(v8bf*)&As[ar][ac + 8] = p1;
    }
    // ---- stage B tile 32x128 transposed: 8 consecutive cols x 2 k-rows ----
    #pragma unroll
    for (int half = 0; half < 2; ++half) {
      int kk = bk + half * 16;        // 0..31
      const float* bp = B + (size_t)(k0 + kk) * N + bn + bc;
      #pragma unroll
      for (int i = 0; i < 8; ++i)
        BsT[bc + i][kk] = f2bf(bp[i]);
    }
    __syncthreads();

    // ---- fragments: both A and B load as 2x ds_load_b128 each ----
    v16bf afrag[2], bfrag[4];
    #pragma unroll
    for (int i = 0; i < 2; ++i)
      afrag[i] = load_frag(&As[wm * 32 + i * 16 + (lane & 15)][0], kb);
    #pragma unroll
    for (int j = 0; j < 4; ++j)
      bfrag[j] = load_frag(&BsT[wn * 64 + j * 16 + (lane & 15)][0], kb);

    // ---- 2x4 WMMA ----
    #pragma unroll
    for (int i = 0; i < 2; ++i)
      #pragma unroll
      for (int j = 0; j < 4; ++j)
        acc[i][j] = __builtin_amdgcn_wmma_f32_16x16x32_bf16(
            false, afrag[i], false, bfrag[j], (short)0, acc[i][j], false, false);
  }

  // ---- epilogue: D layout = N:lane&15, M:vgpr e (+8 for upper half-wave) ----
  #pragma unroll
  for (int i = 0; i < 2; ++i) {
    #pragma unroll
    for (int j = 0; j < 4; ++j) {
      int gm0 = bm + wm * 32 + i * 16 + ((lane >> 4) << 3);
      int gn  = bn + wn * 64 + j * 16 + (lane & 15);
      float bs = bias[gn];
      #pragma unroll
      for (int e = 0; e < 8; ++e) {
        float val = (acc[i][j][e] + bs) * scale;
        if (do_gelu) val = gelu_f(val);
        C[(size_t)(gm0 + e) * N + gn] = val;
      }
    }
  }
}

// ---------------------------------------------------------------------------
// Embedding gather + LayerNorm. One block per token, 256 threads, 3 dims each.
// ---------------------------------------------------------------------------
__global__ __launch_bounds__(256)
void embed_ln(const int* __restrict__ x, const float* __restrict__ wemb,
              const float* __restrict__ pemb, const float* __restrict__ sc,
              const float* __restrict__ bi, float* __restrict__ h)
{
  __shared__ float red[256];
  const int s = blockIdx.x, t = threadIdx.x;
  const int w = s >> 9, p = s & 511;
  const int id = (p == 0) ? 0 : x[w * 511 + p - 1];

  float vals[3];
  float sum = 0.f;
  #pragma unroll
  for (int i = 0; i < 3; ++i) {
    int d = t + 256 * i;
    vals[i] = wemb[(size_t)id * DM_ + d] + pemb[(size_t)s * DM_ + d];
    sum += vals[i];
  }
  red[t] = sum; __syncthreads();
  for (int off = 128; off > 0; off >>= 1) {
    if (t < off) red[t] += red[t + off];
    __syncthreads();
  }
  float mean = red[0] * (1.0f / DM_);
  __syncthreads();
  float vv = 0.f;
  #pragma unroll
  for (int i = 0; i < 3; ++i) { float d = vals[i] - mean; vv += d * d; }
  red[t] = vv; __syncthreads();
  for (int off = 128; off > 0; off >>= 1) {
    if (t < off) red[t] += red[t + off];
    __syncthreads();
  }
  float rstd = rsqrtf(red[0] * (1.0f / DM_) + 1e-5f);
  #pragma unroll
  for (int i = 0; i < 3; ++i) {
    int d = t + 256 * i;
    h[(size_t)s * DM_ + d] = (vals[i] - mean) * rstd * sc[d] + bi[d];
  }
}

// ---------------------------------------------------------------------------
// Residual + LayerNorm in place: h = LN(h + y) * sc + bi
// ---------------------------------------------------------------------------
__global__ __launch_bounds__(256)
void ln_res(float* __restrict__ h, const float* __restrict__ y,
            const float* __restrict__ sc, const float* __restrict__ bi)
{
  __shared__ float red[256];
  const int s = blockIdx.x, t = threadIdx.x;
  float vals[3];
  float sum = 0.f;
  #pragma unroll
  for (int i = 0; i < 3; ++i) {
    int d = t + 256 * i;
    vals[i] = h[(size_t)s * DM_ + d] + y[(size_t)s * DM_ + d];
    sum += vals[i];
  }
  red[t] = sum; __syncthreads();
  for (int off = 128; off > 0; off >>= 1) {
    if (t < off) red[t] += red[t + off];
    __syncthreads();
  }
  float mean = red[0] * (1.0f / DM_);
  __syncthreads();
  float vv = 0.f;
  #pragma unroll
  for (int i = 0; i < 3; ++i) { float d = vals[i] - mean; vv += d * d; }
  red[t] = vv; __syncthreads();
  for (int off = 128; off > 0; off >>= 1) {
    if (t < off) red[t] += red[t + off];
    __syncthreads();
  }
  float rstd = rsqrtf(red[0] * (1.0f / DM_) + 1e-5f);
  #pragma unroll
  for (int i = 0; i < 3; ++i) {
    int d = t + 256 * i;
    h[(size_t)s * DM_ + d] = (vals[i] - mean) * rstd * sc[d] + bi[d];
  }
}

// ---------------------------------------------------------------------------
// Banded local attention + global keys, joint online softmax per query.
// grid: (C*W/64, H), 64 threads (1 query each). q is pre-scaled by 1/sqrt(D).
// ---------------------------------------------------------------------------
__global__ __launch_bounds__(64)
void local_attn(const float* __restrict__ q, const float* __restrict__ k,
                const float* __restrict__ v, float* __restrict__ att)
{
  const int blk = blockIdx.x;          // 0..63
  const int hh  = blockIdx.y;          // head
  const int c   = blk >> 2;            // chunk 0..15
  const int qw  = ((blk & 3) << 6) + threadIdx.x;  // query-in-chunk 0..255
  const int s   = c * W_ + qw;         // global query position

  float qr[64];
  const float* qp = q + (size_t)s * DM_ + hh * D_;
  #pragma unroll
  for (int d = 0; d < 64; ++d) qr[d] = qp[d];

  float m = -1e30f, l = 0.f, o[64];
  #pragma unroll
  for (int d = 0; d < 64; ++d) o[d] = 0.f;

  // 8 global keys (these duplicate in-band keys, as in the reference concat)
  for (int g = 0; g < G_; ++g) {
    int kp = g * (S_ / G_);
    const float* kv = k + (size_t)kp * DM_ + hh * D_;
    const float* vv = v + (size_t)kp * DM_ + hh * D_;
    float sd = 0.f;
    #pragma unroll
    for (int d = 0; d < 64; ++d) sd += qr[d] * kv[d];
    if (sd > m) {
      float r = __expf(m - sd);
      l = l * r + 1.f;
      #pragma unroll
      for (int d = 0; d < 64; ++d) o[d] = o[d] * r + vv[d];
      m = sd;
    } else {
      float pw = __expf(sd - m);
      l += pw;
      #pragma unroll
      for (int d = 0; d < 64; ++d) o[d] += pw * vv[d];
    }
  }
  // local band: j in [qw, qw+2W], kpos = c*W + j - W, valid in [0,S)
  for (int j = qw; j <= qw + 2 * W_; ++j) {
    int kp = c * W_ + j - W_;
    if (kp < 0 || kp >= S_) continue;
    const float* kv = k + (size_t)kp * DM_ + hh * D_;
    const float* vv = v + (size_t)kp * DM_ + hh * D_;
    float sd = 0.f;
    #pragma unroll
    for (int d = 0; d < 64; ++d) sd += qr[d] * kv[d];
    if (sd > m) {
      float r = __expf(m - sd);
      l = l * r + 1.f;
      #pragma unroll
      for (int d = 0; d < 64; ++d) o[d] = o[d] * r + vv[d];
      m = sd;
    } else {
      float pw = __expf(sd - m);
      l += pw;
      #pragma unroll
      for (int d = 0; d < 64; ++d) o[d] += pw * vv[d];
    }
  }
  float inv = 1.f / l;
  float* op = att + (size_t)s * DM_ + hh * D_;
  #pragma unroll
  for (int d = 0; d < 64; ++d) op[d] = o[d] * inv;
}

// ---------------------------------------------------------------------------
// Global tokens: full attention with separate projections, overwrite GPOS rows.
// grid: (G, H), 64 threads. Split-key online softmax, LDS merge.
// ---------------------------------------------------------------------------
__global__ __launch_bounds__(64)
void global_attn(const float* __restrict__ h, const float* __restrict__ Wqg,
                 const float* __restrict__ bqg, const float* __restrict__ kg,
                 const float* __restrict__ vg, float* __restrict__ att)
{
  __shared__ float qs[64], ms[64], ls[64], os[64][65];
  const int g = blockIdx.x, hh = blockIdx.y, t = threadIdx.x;
  const int gpos = g * (S_ / G_);

  // qg[t] = (h[gpos] @ Wqg + bqg)[hh*64+t] * SCALE
  {
    float acc = bqg[hh * D_ + t];
    const float* hrow = h + (size_t)gpos * DM_;
    for (int e = 0; e < DM_; ++e) acc += hrow[e] * Wqg[(size_t)e * DM_ + hh * D_ + t];
    qs[t] = acc * SCALE_;
  }
  __syncthreads();

  float qr[64];
  #pragma unroll
  for (int d = 0; d < 64; ++d) qr[d] = qs[d];

  float m = -1e30f, l = 0.f, o[64];
  #pragma unroll
  for (int d = 0; d < 64; ++d) o[d] = 0.f;

  for (int s = t; s < S_; s += 64) {
    const float* kv = kg + (size_t)s * DM_ + hh * D_;
    const float* vv = vg + (size_t)s * DM_ + hh * D_;
    float sd = 0.f;
    #pragma unroll
    for (int d = 0; d < 64; ++d) sd += qr[d] * kv[d];
    if (sd > m) {
      float r = __expf(m - sd);
      l = l * r + 1.f;
      #pragma unroll
      for (int d = 0; d < 64; ++d) o[d] = o[d] * r + vv[d];
      m = sd;
    } else {
      float pw = __expf(sd - m);
      l += pw;
      #pragma unroll
      for (int d = 0; d < 64; ++d) o[d] += pw * vv[d];
    }
  }
  ms[t] = m; ls[t] = l;
  #pragma unroll
  for (int d = 0; d < 64; ++d) os[t][d] = o[d];
  __syncthreads();

  // merge: thread t owns output dim t
  float M = -1e30f;
  for (int i = 0; i < 64; ++i) M = fmaxf(M, ms[i]);
  float L = 0.f, O = 0.f;
  for (int i = 0; i < 64; ++i) {
    float wgt = __expf(ms[i] - M);
    L += wgt * ls[i];
    O += wgt * os[i][t];
  }
  att[(size_t)gpos * DM_ + hh * D_ + t] = O / L;
}

// ---------------------------------------------------------------------------
// Output head: out[g][o] = h[gpos] @ Wout + bout
// ---------------------------------------------------------------------------
__global__ __launch_bounds__(128)
void head_k(const float* __restrict__ h, const float* __restrict__ Wout,
            const float* __restrict__ bout, float* __restrict__ out)
{
  const int g = blockIdx.x, o = threadIdx.x;
  const float* hr = h + (size_t)(g * (S_ / G_)) * DM_;
  float acc = bout[o];
  for (int d = 0; d < DM_; ++d) acc += hr[d] * Wout[(size_t)d * OUT_ + o];
  out[g * OUT_ + o] = acc;
}

// ---------------------------------------------------------------------------
extern "C" void kernel_launch(void* const* d_in, const int* in_sizes, int n_in,
                              void* d_out, int out_size, void* d_ws, size_t ws_size,
                              hipStream_t stream) {
  (void)in_sizes; (void)n_in; (void)out_size; (void)ws_size;

  const int*   x    = (const int*)  d_in[0];
  const float* wemb = (const float*)d_in[1];
  const float* pemb = (const float*)d_in[2];
  const float* es   = (const float*)d_in[3];
  const float* eb   = (const float*)d_in[4];
  const float* Wq   = (const float*)d_in[5];
  const float* bq   = (const float*)d_in[6];
  const float* Wk   = (const float*)d_in[7];
  const float* bk   = (const float*)d_in[8];
  const float* Wv   = (const float*)d_in[9];
  const float* bv   = (const float*)d_in[10];
  const float* Wo   = (const float*)d_in[11];
  const float* bo   = (const float*)d_in[12];
  const float* Wqg  = (const float*)d_in[13];
  const float* bqg  = (const float*)d_in[14];
  const float* Wkg  = (const float*)d_in[15];
  const float* bkg  = (const float*)d_in[16];
  const float* Wvg  = (const float*)d_in[17];
  const float* bvg  = (const float*)d_in[18];
  const float* ln1s = (const float*)d_in[19];
  const float* ln1b = (const float*)d_in[20];
  const float* W1   = (const float*)d_in[21];
  const float* b1   = (const float*)d_in[22];
  const float* W2   = (const float*)d_in[23];
  const float* b2   = (const float*)d_in[24];
  const float* ln2s = (const float*)d_in[25];
  const float* ln2b = (const float*)d_in[26];
  const float* Wout = (const float*)d_in[27];
  const float* bout = (const float*)d_in[28];

  const size_t SDM = (size_t)S_ * DM_;
  float* h   = (float*)d_ws;
  float* q   = h   + SDM;
  float* k   = q   + SDM;
  float* v   = k   + SDM;
  float* kg  = v   + SDM;
  float* vg  = kg  + SDM;
  float* att = vg  + SDM;
  float* y   = att + SDM;
  float* f1  = y   + SDM;      // S*FF

  embed_ln<<<S_, 256, 0, stream>>>(x, wemb, pemb, es, eb, h);

  const dim3 gP(S_ / 128, DM_ / 128);   // 32 x 6   (N=768)
  const dim3 gF(S_ / 128, FF_ / 128);   // 32 x 24  (N=3072)

  for (int l = 0; l < L_; ++l) {
    const size_t wo = (size_t)l * DM_ * DM_;
    const size_t bo_ = (size_t)l * DM_;
    const size_t w1o = (size_t)l * DM_ * FF_;
    const size_t b1o = (size_t)l * FF_;
    const size_t w2o = (size_t)l * FF_ * DM_;

    gemm_wmma<<<gP, 256, 0, stream>>>(h, Wq  + wo, bq  + bo_, q,  S_, DM_, DM_, SCALE_, 0);
    gemm_wmma<<<gP, 256, 0, stream>>>(h, Wk  + wo, bk  + bo_, k,  S_, DM_, DM_, 1.0f, 0);
    gemm_wmma<<<gP, 256, 0, stream>>>(h, Wv  + wo, bv  + bo_, v,  S_, DM_, DM_, 1.0f, 0);
    gemm_wmma<<<gP, 256, 0, stream>>>(h, Wkg + wo, bkg + bo_, kg, S_, DM_, DM_, 1.0f, 0);
    gemm_wmma<<<gP, 256, 0, stream>>>(h, Wvg + wo, bvg + bo_, vg, S_, DM_, DM_, 1.0f, 0);

    local_attn <<<dim3(C_ * (W_ / 64), H_), 64, 0, stream>>>(q, k, v, att);
    global_attn<<<dim3(G_, H_),             64, 0, stream>>>(h, Wqg + wo, bqg + bo_, kg, vg, att);

    gemm_wmma<<<gP, 256, 0, stream>>>(att, Wo + wo, bo + bo_, y, S_, DM_, DM_, 1.0f, 0);
    ln_res<<<S_, 256, 0, stream>>>(h, y, ln1s + bo_, ln1b + bo_);

    gemm_wmma<<<gF, 256, 0, stream>>>(h,  W1 + w1o, b1 + b1o, f1, S_, FF_, DM_, 1.0f, 1);
    gemm_wmma<<<gP, 256, 0, stream>>>(f1, W2 + w2o, b2 + bo_, y,  S_, DM_, FF_, 1.0f, 0);
    ln_res<<<S_, 256, 0, stream>>>(h, y, ln2s + bo_, ln2b + bo_);
  }

  head_k<<<G_, 128, 0, stream>>>(h, Wout, bout, (float*)d_out);
}
